// Encoder_29867202576597
// MI455X (gfx1250) — matrix operand order repaired and checked by reading, not compiled, not run on hardware
//
#include <hip/hip_runtime.h>

// ---------------------------------------------------------------------------
// LSTM encoder for MI455X (gfx1250, wave32, WMMA).
// B=64, T=256, V=32000, E=512, U=1024, G=4U=4096.
// Strategy: bf16 WMMA (16x16x32, f32 accum) for both the parallel x_gates GEMM
// and the sequential recurrent GEMM; fp32 state/outputs; K-split-4 per step.
// ---------------------------------------------------------------------------

typedef __attribute__((ext_vector_type(16))) __bf16    v16bf;
typedef __attribute__((ext_vector_type(8)))  float     v8f;
typedef __attribute__((ext_vector_type(4)))  unsigned  u32x4;

union Frag { v16bf bf; u32x4 q[2]; };

#define NB 64
#define NT 256
#define NE 512
#define NU 1024
#define NG 4096

__device__ __forceinline__ unsigned short f2bf(float x) {
  unsigned u = __float_as_uint(x);
  return (unsigned short)((u + 0x7FFFu + ((u >> 16) & 1u)) >> 16);  // RNE
}
__device__ __forceinline__ float sigmoidf(float x) {
  return 1.0f / (1.0f + __expf(-x));
}

// ---------------- one-time prep kernels ----------------

__global__ void convert_bf16(const float* __restrict__ src,
                             unsigned short* __restrict__ dst, long n) {
  long i = (long)blockIdx.x * blockDim.x + threadIdx.x;
  if (i < n) dst[i] = f2bf(src[i]);
}

// dst[c*R + r] = bf16(src[r*C + c])   (transpose so WMMA-B reads contiguous K)
__global__ void transpose_bf16(const float* __restrict__ src,
                               unsigned short* __restrict__ dst, int R, int C) {
  long i = (long)blockIdx.x * blockDim.x + threadIdx.x;
  long n = (long)R * C;
  if (i < n) {
    long r = i / C, c = i % C;
    dst[c * (long)R + r] = f2bf(src[i]);
  }
}

__global__ void init_state(const float* __restrict__ h0, const float* __restrict__ c0,
                           unsigned short* __restrict__ hb0, float* __restrict__ cws) {
  int i = blockIdx.x * blockDim.x + threadIdx.x;  // NB*NU = 65536
  hb0[i] = f2bf(h0[i]);
  cws[i] = c0[i];
}

// ---------------- x_gates = gather(emb)[m] @ kernel + bias ----------------
// M = T*B = 16384 (row m -> t=m/64, b=m%64), N = 4096, K = 512.
// Block = 128 threads (4 waves), wave w owns n-tile blockIdx.x*4+w, m-tile blockIdx.y.
__global__ __launch_bounds__(128) void gates_gemm(
    const int* __restrict__ tokens,              // [B][T]
    const unsigned short* __restrict__ embbf,    // [V][E] bf16
    const unsigned short* __restrict__ kT,       // [G][E] bf16 (transposed kernel)
    const float* __restrict__ bias,              // [G]
    float* __restrict__ xg)                      // [T*B][G] fp32
{
  const int lane = threadIdx.x & 31;
  const int wave = threadIdx.x >> 5;
  const int half = lane >> 4;
  const int l16  = lane & 15;
  const int n0 = (blockIdx.x * 4 + wave) * 16;   // 0..4080
  const int m0 = blockIdx.y * 16;                // 0..16368

  // A-fragment row for this lane: gathered embedding row
  const int m   = m0 + l16;
  const int t   = m >> 6;
  const int b   = m & 63;
  const int tok = tokens[b * NT + t];
  const unsigned short* arow = embbf + (long)tok * NE;
  const unsigned short* brow = kT + (long)(n0 + l16) * NE + half * 16;

  v8f acc;
  const float bv = bias[n0 + l16];               // C/D: N = lane&15
#pragma unroll
  for (int r = 0; r < 8; ++r) acc[r] = bv;

  for (int k0 = 0; k0 < NE; k0 += 32) {
    Frag fa, fb;
    fa.q[0] = *(const u32x4*)(arow + k0 + half * 16);       // K: half*16..+8
    fa.q[1] = *(const u32x4*)(arow + k0 + half * 16 + 8);   // K: half*16+8..+8
    fb.q[0] = *(const u32x4*)(brow + k0);                   // K: half*16..+16
    fb.q[1] = *(const u32x4*)(brow + k0 + 8);
    acc = __builtin_amdgcn_wmma_f32_16x16x32_bf16(false, fa.bf, false, fb.bf,
                                                  (short)0, acc, false, false);
  }

#pragma unroll
  for (int r = 0; r < 8; ++r) {
    int Mr = r + 8 * half;                        // C/D layout
    xg[(long)(m0 + Mr) * NG + n0 + l16] = acc[r];
  }
}

// ---------------- one recurrent timestep ----------------
// z = xg[t] + h @ rec_kernel ; gates ; c,h update.
// Grid (U/16=64, B/16=4), block 128 = 4 waves splitting K=1024 four ways.
__global__ __launch_bounds__(128) void lstm_step(
    const float* __restrict__ xg_t,              // xg + t*B*G
    const unsigned short* __restrict__ rkT,      // [G][U] bf16 (transposed rec_kernel)
    const unsigned short* __restrict__ hin,      // [B][U] bf16
    unsigned short* __restrict__ hout,           // [B][U] bf16
    float* __restrict__ cws,                     // [B][U] fp32
    float* __restrict__ out_t,                   // out + t*U  (stride T*U per b-row)
    float* __restrict__ hlast, float* __restrict__ clast,
    int is_last)
{
  __shared__ float red[3][4][256];               // partials from waves 1..3
  const int lane = threadIdx.x & 31;
  const int wave = threadIdx.x >> 5;
  const int half = lane >> 4;
  const int l16  = lane & 15;
  const int u0 = blockIdx.x * 16;                // 0..1008
  const int m0 = blockIdx.y * 16;                // 0..48
  const int kbase = wave * 256;                  // K split across 4 waves

  v8f acc[4];
#pragma unroll
  for (int x = 0; x < 4; ++x)
#pragma unroll
    for (int r = 0; r < 8; ++r) acc[x][r] = 0.0f;

  const unsigned short* arow = hin + (long)(m0 + l16) * NU;
  const unsigned short* brow[4];
#pragma unroll
  for (int x = 0; x < 4; ++x)
    brow[x] = rkT + (long)(x * NU + u0 + l16) * NU + half * 16;

  for (int k0 = kbase; k0 < kbase + 256; k0 += 32) {
    Frag fa;
    fa.q[0] = *(const u32x4*)(arow + k0 + half * 16);
    fa.q[1] = *(const u32x4*)(arow + k0 + half * 16 + 8);
#pragma unroll
    for (int x = 0; x < 4; ++x) {
      const unsigned short* bp = brow[x] + k0;
      __builtin_prefetch((const void*)(bp + 64), 0, 3);     // global_prefetch_b8
      Frag fb;
      fb.q[0] = *(const u32x4*)(bp);
      fb.q[1] = *(const u32x4*)(bp + 8);
      acc[x] = __builtin_amdgcn_wmma_f32_16x16x32_bf16(false, fa.bf, false, fb.bf,
                                                       (short)0, acc[x], false, false);
    }
  }

  if (wave != 0) {
#pragma unroll
    for (int x = 0; x < 4; ++x)
#pragma unroll
      for (int r = 0; r < 8; ++r)
        red[wave - 1][x][r * 32 + lane] = acc[x][r];
  }
  __syncthreads();

  if (wave == 0) {
#pragma unroll
    for (int r = 0; r < 8; ++r) {
      const int Mr = r + 8 * half;               // C/D layout: M
      const int row = m0 + Mr;                   // b index 0..63
      const int col = u0 + l16;                  // u index 0..1023
      float z[4];
#pragma unroll
      for (int x = 0; x < 4; ++x) {
        float s = acc[x][r]
                + red[0][x][r * 32 + lane]
                + red[1][x][r * 32 + lane]
                + red[2][x][r * 32 + lane];
        s += xg_t[(long)row * NG + x * NU + col];
        z[x] = s;
      }
      const float iv = sigmoidf(z[0]);
      const float fv = sigmoidf(z[1]);
      const float gv = sigmoidf(z[2]);           // reference uses sigmoid, not tanh
      const float ov = sigmoidf(z[3]);
      const float cn = fv * cws[row * NU + col] + iv * gv;
      const float hn = ov * sigmoidf(cn);
      cws[row * NU + col] = cn;
      out_t[(long)row * (NT * NU) + col] = hn;
      hout[row * NU + col] = f2bf(hn);
      if (is_last) {
        hlast[row * NU + col] = hn;
        clast[row * NU + col] = cn;
      }
    }
  }
}

// ---------------- host launcher ----------------

extern "C" void kernel_launch(void* const* d_in, const int* in_sizes, int n_in,
                              void* d_out, int out_size, void* d_ws, size_t ws_size,
                              hipStream_t stream) {
  (void)in_sizes; (void)n_in; (void)out_size; (void)ws_size;
  const int*   tokens = (const int*)  d_in[0];   // [B][T]
  const float* h0     = (const float*)d_in[1];   // [B][U]
  const float* c0     = (const float*)d_in[2];   // [B][U]
  const float* emb    = (const float*)d_in[3];   // [V][E]
  const float* kern   = (const float*)d_in[4];   // [E][G]
  const float* reck   = (const float*)d_in[5];   // [U][G]
  const float* bias   = (const float*)d_in[6];   // [G]

  float* out   = (float*)d_out;                  // [B][T][U]
  float* hlast = out + (size_t)NB * NT * NU;     // [B][U]
  float* clast = hlast + (size_t)NB * NU;        // [B][U]

  char* ws = (char*)d_ws;
  size_t off = 0;
  float*          xg    = (float*)(ws + off);          off += (size_t)NT * NB * NG * 4;   // 268.4 MB
  unsigned short* embbf = (unsigned short*)(ws + off); off += (size_t)32000 * NE * 2;     // 32.8 MB
  unsigned short* kT    = (unsigned short*)(ws + off); off += (size_t)NG * NE * 2;        // 4.2 MB
  unsigned short* rkT   = (unsigned short*)(ws + off); off += (size_t)NG * NU * 2;        // 8.4 MB
  unsigned short* hb0   = (unsigned short*)(ws + off); off += (size_t)NB * NU * 2;
  unsigned short* hb1   = (unsigned short*)(ws + off); off += (size_t)NB * NU * 2;
  float*          cws   = (float*)(ws + off);          off += (size_t)NB * NU * 4;

  // 1) precision conversion / layout transforms
  {
    long n = (long)32000 * NE;
    convert_bf16<<<(unsigned)((n + 255) / 256), 256, 0, stream>>>(emb, embbf, n);
  }
  {
    long n = (long)NE * NG;
    transpose_bf16<<<(unsigned)((n + 255) / 256), 256, 0, stream>>>(kern, kT, NE, NG);
  }
  {
    long n = (long)NU * NG;
    transpose_bf16<<<(unsigned)((n + 255) / 256), 256, 0, stream>>>(reck, rkT, NU, NG);
  }
  init_state<<<(NB * NU) / 256, 256, 0, stream>>>(h0, c0, hb0, cws);

  // 2) parallel x_gates GEMM: grid = (Ntiles/4, Mtiles) = (64, 1024)
  gates_gemm<<<dim3(NG / 64, (NT * NB) / 16), 128, 0, stream>>>(tokens, embbf, kT, bias, xg);

  // 3) sequential scan: 256 fused GEMM+cell steps
  for (int t = 0; t < NT; ++t) {
    const unsigned short* hin  = (t & 1) ? hb1 : hb0;
    unsigned short*       hout = (t & 1) ? hb0 : hb1;
    lstm_step<<<dim3(NU / 16, NB / 16), 128, 0, stream>>>(
        xg + (size_t)t * NB * NG, rkT, hin, hout, cws,
        out + (size_t)t * NU, hlast, clast, (t == NT - 1) ? 1 : 0);
  }
}
